// GAT_81509889344205
// MI455X (gfx1250) — matrix operand order repaired and checked by reading, not compiled, not run on hardware
//
#include <hip/hip_runtime.h>

typedef __attribute__((ext_vector_type(16))) __bf16        v16bf;
typedef __attribute__((ext_vector_type(8)))  float         v8f;
typedef __attribute__((ext_vector_type(8)))  unsigned int  v8u;
typedef __attribute__((ext_vector_type(4)))  float         f32x4;
typedef __attribute__((ext_vector_type(2)))  unsigned int  u32x2;

// ---------------------------------------------------------------- helpers
static __device__ __forceinline__ unsigned short f2bf(float f) {
  unsigned int u = __builtin_bit_cast(unsigned int, f);
  u += 0x7FFFu + ((u >> 16) & 1u);          // round-to-nearest-even
  return (unsigned short)(u >> 16);
}
static __device__ __forceinline__ unsigned int pk2bf(float lo, float hi) {
  return (unsigned int)f2bf(lo) | ((unsigned int)f2bf(hi) << 16);
}

// ---------------------------------------------------------------- WMMA GEMM
// C[M,N] = A[M,K] * B[K,N], fp32 in/out, bf16 multiply / f32 accumulate.
// Block = 256 threads (8 waves). C tile 64x32 (4x2 waves of 16x16 WMMA).
// BK = 64 -> 2 WMMA per wave per barrier pair.
// KTAIL=false requires K % 64 == 0 (aligned float4 fast path).
// Requires: N % 32 == 0 (pad B), M row clamping makes all loads in-bounds.
template <bool KTAIL>
__global__ __launch_bounds__(256)
void gemm_bf16_wmma(const float* __restrict__ A, const float* __restrict__ B,
                    float* __restrict__ C, int M, int N, int K,
                    int lda, int ldb, int ldc)
{
  constexpr int BM = 64, BN = 32, BK = 64;
  __shared__ unsigned short As[BM][BK + 8];   // stride 72 ushorts = 144 B
  __shared__ unsigned short Bs[BN][BK + 8];   // transposed: [n][k]

  const int tid   = threadIdx.x;
  const int lane  = tid & 31;
  const int wave  = tid >> 5;
  const int wm    = (wave & 3) * 16;          // wave M offset in tile
  const int wn    = (wave >> 2) * 16;         // wave N offset in tile
  const int tileM = blockIdx.y * BM;
  const int tileN = blockIdx.x * BN;

  v8f acc = {};

  for (int k0 = 0; k0 < K; k0 += BK) {
    // ---- stage A tile (64x64), 4 floats / thread / iter, branchless
#pragma unroll
    for (int i = tid * 4; i < BM * BK; i += 256 * 4) {
      const int r = i >> 6, c = i & (BK - 1);
      const int gr = min(tileM + r, M - 1);                   // clamp: in-bounds
      const float* ap = A + (size_t)gr * lda;
      float a0, a1, a2, a3;
      if (KTAIL) {
        const int k = k0 + c;
        a0 = ap[min(k,     K - 1)]; a0 = (k     < K) ? a0 : 0.f;
        a1 = ap[min(k + 1, K - 1)]; a1 = (k + 1 < K) ? a1 : 0.f;
        a2 = ap[min(k + 2, K - 1)]; a2 = (k + 2 < K) ? a2 : 0.f;
        a3 = ap[min(k + 3, K - 1)]; a3 = (k + 3 < K) ? a3 : 0.f;
      } else {
        f32x4 v = *(const f32x4*)(ap + k0 + c);               // aligned b128
        a0 = v.x; a1 = v.y; a2 = v.z; a3 = v.w;
      }
      u32x2 p; p.x = pk2bf(a0, a1); p.y = pk2bf(a2, a3);
      *(u32x2*)&As[r][c] = p;                                 // 8B-aligned
    }
    // ---- stage B tile (64x32) transposed, 4 floats / thread / iter
#pragma unroll
    for (int i = tid * 4; i < BK * BN; i += 256 * 4) {
      const int r = i >> 5, c = i & (BN - 1);                 // r = k, c = n
      const int gk = k0 + r;
      const float* bp = B + (size_t)(KTAIL ? min(gk, K - 1) : gk) * ldb + tileN + c;
      f32x4 v = *(const f32x4*)bp;                            // aligned b128
      if (KTAIL) { const float m = (gk < K) ? 1.f : 0.f; v *= m; }
      Bs[c + 0][r] = f2bf(v.x);
      Bs[c + 1][r] = f2bf(v.y);
      Bs[c + 2][r] = f2bf(v.z);
      Bs[c + 3][r] = f2bf(v.w);
    }
    __syncthreads();

#pragma unroll
    for (int kc = 0; kc < BK; kc += 32) {
      // A fragment: lane L -> row L%16; lanes>=16 shift K by 16.
      const int row = wm + (lane & 15);
      const int kg  = kc + (lane >> 4) * 16;
      v8u au;
#pragma unroll
      for (int v = 0; v < 4; ++v) {
        au[v]     = *(const unsigned int*)&As[row][kg + 2 * v];
        au[v + 4] = *(const unsigned int*)&As[row][kg + 8 + 2 * v];
      }
      // B fragment: lane L -> col L%16; lanes>=16 K+=16; VGPR v = K pair 2v.
      const int col = wn + (lane & 15);
      const int kgb = kc + (lane >> 4) * 16;
      v8u bu;
#pragma unroll
      for (int v = 0; v < 8; ++v)
        bu[v] = *(const unsigned int*)&Bs[col][kgb + 2 * v];

      acc = __builtin_amdgcn_wmma_f32_16x16x32_bf16(
          false, __builtin_bit_cast(v16bf, au),
          false, __builtin_bit_cast(v16bf, bu),
          (short)0, acc, false, false);
    }
    __syncthreads();
  }

  // D layout: VGPR r -> M = r + 8*(lane>=16), N = lane%16
  const int cn = tileN + wn + (lane & 15);
  const int mb = tileM + wm + (lane >> 4) * 8;
#pragma unroll
  for (int r = 0; r < 8; ++r) {
    const int m = mb + r;
    if (m < M) C[(size_t)m * ldc + cn] = acc[r];
  }
}

// ---------------------------------------------------------------- fold We @ att_edge
// Wef[d, l*8+h] = sum_c We_l[d, h*32+c] * ae_l[h*32+c]  -> padded [160][32]
__global__ void fold_we_kernel(const float* __restrict__ We0, const float* __restrict__ ae0,
                               const float* __restrict__ We1, const float* __restrict__ ae1,
                               const float* __restrict__ We2, const float* __restrict__ ae2,
                               float* __restrict__ Wef)
{
  int idx = blockIdx.x * blockDim.x + threadIdx.x;
  if (idx >= 147 * 24) return;
  int d = idx / 24, col = idx % 24;
  int l = col >> 3, hh = col & 7;
  const float* We = (l == 0) ? We0 : (l == 1) ? We1 : We2;
  const float* ae = (l == 0) ? ae0 : (l == 1) ? ae1 : ae2;
  float s = 0.f;
#pragma unroll 8
  for (int c = 0; c < 32; ++c)
    s += We[(size_t)d * 256 + hh * 32 + c] * ae[hh * 32 + c];
  Wef[(size_t)d * 32 + col] = s;
}

// ---------------------------------------------------------------- alpha_src / alpha_dst
__global__ void node_alpha_kernel(const float* __restrict__ h,
                                  const float* __restrict__ as,
                                  const float* __restrict__ ad,
                                  float* __restrict__ alpha_s,
                                  float* __restrict__ alpha_d, int n)
{
  int idx = blockIdx.x * blockDim.x + threadIdx.x;
  if (idx >= n * 8) return;
  int node = idx >> 3, hh = idx & 7;
  const float* hp = h + (size_t)node * 256 + hh * 32;
  float ss = 0.f, sd = 0.f;
#pragma unroll 8
  for (int c = 0; c < 32; ++c) {
    float v = hp[c];
    ss += v * as[hh * 32 + c];
    sd += v * ad[hh * 32 + c];
  }
  alpha_s[idx] = ss;
  alpha_d[idx] = sd;
}

// ---------------------------------------------------------------- pass 1: logits + segment max
__global__ void edge_logits_kernel(const int* __restrict__ src, const int* __restrict__ dst,
                                   const float* __restrict__ alpha_s,
                                   const float* __restrict__ alpha_d,
                                   const float* __restrict__ alpha_e,  // [E,32]
                                   int layer, float* __restrict__ logits,
                                   unsigned int* __restrict__ mkey, int ne)
{
  int idx = blockIdx.x * blockDim.x + threadIdx.x;
  if (idx >= ne * 8) return;
  int e = idx >> 3, hh = idx & 7;
  int s = src[e], d = dst[e];
  float lg = alpha_s[s * 8 + hh] + alpha_d[d * 8 + hh]
           + alpha_e[(size_t)e * 32 + layer * 8 + hh];
  lg = (lg >= 0.f) ? lg : 0.2f * lg;           // leaky relu
  logits[idx] = lg;
  unsigned int ub  = __builtin_bit_cast(unsigned int, lg);
  unsigned int key = ((int)ub >= 0) ? (ub | 0x80000000u) : ~ub;  // order-preserving
  atomicMax(&mkey[d * 8 + hh], key);
}

// ---------------------------------------------------------------- pass 2: exp + segment sum
__global__ void edge_exp_kernel(const int* __restrict__ dst,
                                const unsigned int* __restrict__ mkey,
                                float* __restrict__ logits,
                                float* __restrict__ denom, int ne)
{
  int idx = blockIdx.x * blockDim.x + threadIdx.x;
  if (idx >= ne * 8) return;
  int e = idx >> 3, hh = idx & 7;
  int d = dst[e];
  unsigned int key = mkey[d * 8 + hh];
  unsigned int ub  = (key & 0x80000000u) ? (key & 0x7FFFFFFFu) : ~key;
  float m  = __builtin_bit_cast(float, ub);
  float ex = __expf(logits[idx] - m);
  logits[idx] = ex;                             // reuse buffer for exp values
  atomicAdd(&denom[d * 8 + hh], ex);
}

// ---------------------------------------------------------------- pass 3: weighted scatter
// 4 features per thread: one b128 gather of h[src], 4 f32 atomic adds.
__global__ void edge_msg_kernel(const int* __restrict__ src, const int* __restrict__ dst,
                                const float* __restrict__ h,
                                const float* __restrict__ ex,
                                const float* __restrict__ denom,
                                float* __restrict__ agg, int ne)
{
  int gid = blockIdx.x * blockDim.x + threadIdx.x;
  if (gid >= ne * 64) return;
  int e = gid >> 6, q = gid & 63;
  int f0 = q << 2, hh = q >> 3;
  int s = src[e], d = dst[e];
  float a = ex[(size_t)e * 8 + hh] / (denom[(size_t)d * 8 + hh] + 1e-16f);
  f32x4 hv = *(const f32x4*)&h[(size_t)s * 256 + f0];
  float* op = &agg[(size_t)d * 256 + f0];
  atomicAdd(op + 0, hv.x * a);
  atomicAdd(op + 1, hv.y * a);
  atomicAdd(op + 2, hv.z * a);
  atomicAdd(op + 3, hv.w * a);
}

// ---------------------------------------------------------------- bias + (optional) ELU
__global__ void bias_act_kernel(const float* __restrict__ agg, const float* __restrict__ b,
                                float* __restrict__ out, int n, int do_elu)
{
  int idx = blockIdx.x * blockDim.x + threadIdx.x;
  if (idx >= n * 256) return;
  int f = idx & 255;
  float v = agg[idx] + b[f];
  if (do_elu) v = (v > 0.f) ? v : (__expf(v) - 1.0f);
  out[idx] = v;
}

// ---------------------------------------------------------------- mean pool (atomics)
__global__ void pool_kernel(const float* __restrict__ h, const int* __restrict__ batch,
                            float* __restrict__ pooled, float* __restrict__ cnt, int n)
{
  int idx = blockIdx.x * blockDim.x + threadIdx.x;
  if (idx >= n * 256) return;
  int node = idx >> 8, f = idx & 255;
  int g = batch[node];
  atomicAdd(&pooled[(size_t)g * 256 + f], h[idx]);
  if (f == 0) atomicAdd(&cnt[g], 1.0f);
}

// ---------------------------------------------------------------- classifier
__global__ void classifier_kernel(const float* __restrict__ pooled,
                                  const float* __restrict__ cnt,
                                  const float* __restrict__ Wc,
                                  const float* __restrict__ bc,
                                  float* __restrict__ out)
{
  int idx = blockIdx.x * blockDim.x + threadIdx.x;
  if (idx >= 64 * 32) return;
  int b = idx >> 5, c = idx & 31;
  float inv = 1.0f / fmaxf(cnt[b], 1.0f);
  float s = bc[c];
  for (int k = 0; k < 256; ++k)
    s += pooled[(size_t)b * 256 + k] * inv * Wc[(size_t)k * 32 + c];
  out[idx] = s;
}

// ================================================================ launcher
extern "C" void kernel_launch(void* const* d_in, const int* in_sizes, int n_in,
                              void* d_out, int out_size, void* d_ws, size_t ws_size,
                              hipStream_t stream)
{
  (void)in_sizes; (void)n_in; (void)out_size; (void)ws_size;
  const int N_ = 20000, E_ = 320000, HC = 256, ED_ = 147, B_ = 64;

  const float* x        = (const float*)d_in[0];
  const int*   eidx     = (const int*)d_in[1];
  const int*   src      = eidx;
  const int*   dst      = eidx + E_;
  const float* edgeattr = (const float*)d_in[2];
  const int*   batch    = (const int*)d_in[3];
  const float* Wc       = (const float*)d_in[22];
  const float* bc       = (const float*)d_in[23];

  // workspace carve-up (floats)
  float* w = (float*)d_ws;
  float* hA     = w;  w += (size_t)N_ * HC;     // transformed features
  float* hB     = w;  w += (size_t)N_ * HC;     // layer output
  float* agg    = w;  w += (size_t)N_ * HC;     // message accumulator
  float* alphaE = w;  w += (size_t)E_ * 32;     // fused 3-layer edge logits (ld 32)
  float* Wef    = w;  w += 160 * 32;            // zero-padded [160][32]
  float* aS     = w;  w += (size_t)N_ * 8;
  float* aD     = w;  w += (size_t)N_ * 8;
  float* lg     = w;  w += (size_t)E_ * 8;      // logits, then exp values
  unsigned int* mk = (unsigned int*)w; w += (size_t)N_ * 8;
  float* den    = w;  w += (size_t)N_ * 8;
  float* pooled = w;  w += (size_t)B_ * HC;
  float* cnt    = w;  w += B_;

  // 1. fold per-layer (We, att_edge) into zero-padded [160][32]
  hipMemsetAsync(Wef, 0, 160 * 32 * 4, stream);
  fold_we_kernel<<<(147 * 24 + 255) / 256, 256, 0, stream>>>(
      (const float*)d_in[8],  (const float*)d_in[7],
      (const float*)d_in[14], (const float*)d_in[13],
      (const float*)d_in[20], (const float*)d_in[19], Wef);

  // 2. alpha_e for all 3 layers in one pass over edge_attr: [E,147]@[160p,32]
  {
    dim3 grid(1, (E_ + 63) / 64);
    gemm_bf16_wmma<true><<<grid, 256, 0, stream>>>(edgeattr, Wef, alphaE,
                                                   E_, 32, ED_, ED_, 32, 32);
  }

  const float* hin = x;
  int Fin = 32;
  for (int l = 0; l < 3; ++l) {
    const float* Wl = (const float*)d_in[4 + 6 * l];
    const float* as = (const float*)d_in[5 + 6 * l];
    const float* ad = (const float*)d_in[6 + 6 * l];
    const float* bl = (const float*)d_in[9 + 6 * l];

    // node transform h = hin @ Wl  (bf16 WMMA)
    {
      dim3 grid(HC / 32, (N_ + 63) / 64);
      if (Fin % 64 == 0)
        gemm_bf16_wmma<false><<<grid, 256, 0, stream>>>(hin, Wl, hA,
                                                        N_, HC, Fin, Fin, HC, HC);
      else
        gemm_bf16_wmma<true><<<grid, 256, 0, stream>>>(hin, Wl, hA,
                                                       N_, HC, Fin, Fin, HC, HC);
    }

    node_alpha_kernel<<<(N_ * 8 + 255) / 256, 256, 0, stream>>>(hA, as, ad, aS, aD, N_);

    hipMemsetAsync(mk,  0, (size_t)N_ * 8 * 4, stream);
    hipMemsetAsync(den, 0, (size_t)N_ * 8 * 4, stream);
    hipMemsetAsync(agg, 0, (size_t)N_ * HC * 4, stream);

    edge_logits_kernel<<<(E_ * 8 + 255) / 256, 256, 0, stream>>>(
        src, dst, aS, aD, alphaE, l, lg, mk, E_);
    edge_exp_kernel<<<(E_ * 8 + 255) / 256, 256, 0, stream>>>(dst, mk, lg, den, E_);
    edge_msg_kernel<<<(E_ * 64 + 255) / 256, 256, 0, stream>>>(
        src, dst, hA, lg, den, agg, E_);

    bias_act_kernel<<<((size_t)N_ * HC + 255) / 256, 256, 0, stream>>>(
        agg, bl, hB, N_, (l < 2) ? 1 : 0);

    hin = hB;
    Fin = HC;
  }

  // mean pool + classifier
  hipMemsetAsync(pooled, 0, (size_t)B_ * HC * 4, stream);
  hipMemsetAsync(cnt,    0, (size_t)B_ * 4, stream);
  pool_kernel<<<((size_t)N_ * HC + 255) / 256, 256, 0, stream>>>(hB, batch, pooled, cnt, N_);
  classifier_kernel<<<(64 * 32 + 255) / 256, 256, 0, stream>>>(
      pooled, cnt, Wc, bc, (float*)d_out);
}
